// DAM_50689204027803
// MI455X (gfx1250) — matrix operand order repaired and checked by reading, not compiled, not run on hardware
//
#include <hip/hip_runtime.h>
#include <hip/hip_bf16.h>

typedef __attribute__((ext_vector_type(16))) _Float16 v16h;
typedef __attribute__((ext_vector_type(8)))  _Float16 v8h;
typedef __attribute__((ext_vector_type(8)))  float    v8f;

#define DEV static __device__ __forceinline__

DEV v8f vzero() {
    v8f z;
#pragma unroll
    for (int i = 0; i < 8; ++i) z[i] = 0.f;
    return z;
}

// A-operand fragment, 16x32 f16 (row-major source, row = M).
// Per CDNA5 ISA 7.12.2: lane L(0-15)=M, lane group g=L>>4 selects K-halves:
//   elements 0..7  <-> K = g*8 + 0..7
//   elements 8..15 <-> K = 16 + g*8 + 0..7
DEV v16h frag_a(const _Float16* row, int g) {
    const v8h lo = *(const v8h*)(row + g * 8);
    const v8h hi = *(const v8h*)(row + 16 + g * 8);
    v16h r;
#pragma unroll
    for (int i = 0; i < 8; ++i) { r[i] = lo[i]; r[i + 8] = hi[i]; }
    return r;
}

// B-operand fragment, 32x16 f16. B stored transposed as Bst[n][k] (n-major).
// Same K layout as A: elements 0..7 <-> K = g*8 + 0..7,
// elements 8..15 <-> K = 16 + g*8 + 0..7.
DEV v16h frag_b(const _Float16* row, int g) {
    const v8h lo = *(const v8h*)(row + g * 8);
    const v8h hi = *(const v8h*)(row + 16 + g * 8);
    v16h r;
#pragma unroll
    for (int i = 0; i < 8; ++i) { r[i] = lo[i]; r[i + 8] = hi[i]; }
    return r;
}

DEV v8f wmma16(v16h a, v16h b, v8f c) {
    return __builtin_amdgcn_wmma_f32_16x16x32_f16(false, a, false, b, (short)0, c, false, false);
}

// ---------------------------------------------------------------- small utils
__global__ void k_bn_prep(const float* __restrict__ g, const float* __restrict__ b,
                          const float* __restrict__ m, const float* __restrict__ v,
                          float* __restrict__ scale, float* __restrict__ shift) {
    int i = threadIdx.x;  // 128
    float s = g[i] * rsqrtf(v[i] + 1e-5f);
    scale[i] = s;
    shift[i] = b[i] - m[i] * s;
}

__global__ void k_pack_w3(const float* __restrict__ w, _Float16* __restrict__ wp, int Ci) {
    int n = 128 * Ci * 9;
    for (int idx = blockIdx.x * blockDim.x + threadIdx.x; idx < n; idx += gridDim.x * blockDim.x) {
        int r = idx % 9;
        int t = idx / 9;
        int ci = t % Ci;
        int co = t / Ci;
        wp[(size_t)co * 9 * Ci + (size_t)r * Ci + ci] = (_Float16)w[idx];
    }
}

__global__ void k_cvt_h(const float* __restrict__ s, _Float16* __restrict__ d, int n) {
    for (int i = blockIdx.x * blockDim.x + threadIdx.x; i < n; i += gridDim.x * blockDim.x)
        d[i] = (_Float16)s[i];
}

__global__ void k_zero_h(_Float16* __restrict__ p, int n) {
    for (int i = blockIdx.x * blockDim.x + threadIdx.x; i < n; i += gridDim.x * blockDim.x)
        p[i] = (_Float16)0.f;
}

__global__ void k_sum_h(const _Float16* __restrict__ a, const _Float16* __restrict__ b,
                        _Float16* __restrict__ d, int n) {
    for (int i = blockIdx.x * blockDim.x + threadIdx.x; i < n; i += gridDim.x * blockDim.x)
        d[i] = (_Float16)((float)a[i] + (float)b[i]);
}

// -------------------------------------------- 3x3 conv (implicit GEMM, WMMA)
// out[co, pix] = sum_{r,ci} wp[co][r*CI+ci] * x[b][ci][h+kh-1][w+kw-1], BN+ReLU.
// M = 128 (one block covers all co), N-tile = 128 pixels (2 image rows).
template <int CI, typename TIN>
__global__ void __launch_bounds__(256) k_conv3x3(
    const TIN* __restrict__ xin,          // [B][CI][64][64]
    const _Float16* __restrict__ wp,      // [128][9*CI]
    const float* __restrict__ scale, const float* __restrict__ shift,
    float* __restrict__ out32,            // [B][128][4096] or null
    _Float16* __restrict__ outh,          // [B][128][4096] or null
    _Float16* __restrict__ outT) {        // [B][4096][128] or null
    constexpr int KT = 9 * CI;
    __shared__ _Float16 Bst[128 * 40];    // Bst[pixel][k], row stride 40 halves (80B, 16B-aligned)
    const int tid = threadIdx.x;
    const int wv = tid >> 5, lane = tid & 31, nl = lane & 15, g = lane >> 4;
    const int b = blockIdx.y;
    const int pixbase = blockIdx.x * 128;
    const int h0 = pixbase >> 6;

    v8f acc[8];
#pragma unroll
    for (int nt = 0; nt < 8; ++nt) acc[nt] = vzero();

    const int ldci = tid >> 3;        // 0..31: k within chunk
    const int ldp0 = (tid & 7) * 16;  // pixel start

    for (int r = 0; r < 9; ++r) {
        const int kh = r / 3, kw = r % 3;
        for (int c0 = 0; c0 < CI; c0 += 32) {
            __syncthreads();
            {   // stage B tile: Bst[p][k] = x[b][c0+ldci][h0 + p/64 + kh-1][p%64 + kw-1]
                const TIN* src = xin + ((size_t)(b * CI + c0 + ldci) << 12);
#pragma unroll
                for (int i = 0; i < 16; ++i) {
                    int p = ldp0 + i;
                    int hh = h0 + (p >> 6) + kh - 1;
                    int ww = (p & 63) + kw - 1;
                    _Float16 vv = (_Float16)0.f;
                    if ((unsigned)hh < 64u && (unsigned)ww < 64u) vv = (_Float16)src[(hh << 6) + ww];
                    Bst[p * 40 + ldci] = vv;
                }
            }
            __syncthreads();
            const int k0 = r * CI + c0;
            const _Float16* arow = wp + (size_t)(wv * 16 + nl) * KT + k0;
            __builtin_prefetch((const void*)(arow + 32), 0, 1);   // global_prefetch_b8
            v16h a = frag_a(arow, g);
#pragma unroll
            for (int nt = 0; nt < 8; ++nt) {
                v16h bb = frag_b(&Bst[(nt * 16 + nl) * 40], g);
                acc[nt] = wmma16(a, bb, acc[nt]);
            }
        }
    }
#pragma unroll
    for (int nt = 0; nt < 8; ++nt) {
#pragma unroll
        for (int i = 0; i < 8; ++i) {
            int co = wv * 16 + g * 8 + i;            // C-layout: lanes16-31 -> M+8
            int pix = pixbase + nt * 16 + nl;
            float val = acc[nt][i] * scale[co] + shift[co];
            val = val > 0.f ? val : 0.f;
            size_t idx = ((size_t)(b * 128 + co) << 12) + pix;
            if (out32) out32[idx] = val;
            if (outh) outh[idx] = (_Float16)val;
            if (outT) outT[(((size_t)b << 12) + pix) * 128 + co] = (_Float16)val;
        }
    }
}

// ------------------------------------------ generic M=128 GEMM: A[128][K] x Bst[N][K]
__global__ void __launch_bounds__(256) k_gemm128(
    const _Float16* __restrict__ A, size_t aBatchStride,
    const _Float16* __restrict__ Bst,      // [B][4096][K]
    int K,
    const float* __restrict__ bias,        // [128] or null
    const float* __restrict__ gamma,       // scalar or null
    const float* __restrict__ resid,       // [B][128][4096] or null
    _Float16* __restrict__ outh) {         // [B][128][4096]
    const int tid = threadIdx.x, wv = tid >> 5, lane = tid & 31, nl = lane & 15, g = lane >> 4;
    const int b = blockIdx.y;
    const int pixbase = blockIdx.x * 128;
    const _Float16* Ab = A + (size_t)b * aBatchStride;
    const _Float16* Bb = Bst + ((size_t)b << 12) * K;
    v8f acc[8];
#pragma unroll
    for (int nt = 0; nt < 8; ++nt) acc[nt] = vzero();
    for (int k0 = 0; k0 < K; k0 += 32) {
        v16h a = frag_a(Ab + (size_t)(wv * 16 + nl) * K + k0, g);
#pragma unroll
        for (int nt = 0; nt < 8; ++nt) {
            v16h bb = frag_b(Bb + (size_t)(pixbase + nt * 16 + nl) * K + k0, g);
            acc[nt] = wmma16(a, bb, acc[nt]);
        }
    }
    const float gm = gamma ? gamma[0] : 1.f;
#pragma unroll
    for (int nt = 0; nt < 8; ++nt) {
#pragma unroll
        for (int i = 0; i < 8; ++i) {
            int c = wv * 16 + g * 8 + i;
            int pix = pixbase + nt * 16 + nl;
            size_t idx = ((size_t)(b * 128 + c) << 12) + pix;
            float val = gm * acc[nt][i];
            if (bias) val += bias[c];
            if (resid) val += resid[idx];
            outh[idx] = (_Float16)val;
        }
    }
}

// ------------------------------------------ M=64 GEMM (1x1 output convs) -> f32 d_out
__global__ void __launch_bounds__(256) k_gemm64(
    const _Float16* __restrict__ A,       // [64][128]
    const _Float16* __restrict__ BT,      // [B][4096][128]
    const float* __restrict__ bias,       // [64]
    float* __restrict__ out) {            // [B][64][4096]
    const int tid = threadIdx.x, wv = tid >> 5, lane = tid & 31, nl = lane & 15, g = lane >> 4;
    const int mt = wv & 3, nh = wv >> 2;
    const int b = blockIdx.y;
    const int pixbase = blockIdx.x * 128 + nh * 64;
    const _Float16* Bb = BT + ((size_t)b << 12) * 128;
    v8f acc[4];
#pragma unroll
    for (int nt = 0; nt < 4; ++nt) acc[nt] = vzero();
    for (int k0 = 0; k0 < 128; k0 += 32) {
        v16h a = frag_a(A + (size_t)(mt * 16 + nl) * 128 + k0, g);
#pragma unroll
        for (int nt = 0; nt < 4; ++nt) {
            v16h bb = frag_b(Bb + (size_t)(pixbase + nt * 16 + nl) * 128 + k0, g);
            acc[nt] = wmma16(a, bb, acc[nt]);
        }
    }
#pragma unroll
    for (int nt = 0; nt < 4; ++nt) {
#pragma unroll
        for (int i = 0; i < 8; ++i) {
            int co = mt * 16 + g * 8 + i;
            int pix = pixbase + nt * 16 + nl;
            out[(((size_t)b * 64 + co) << 12) + pix] = acc[nt][i] + bias[co];
        }
    }
}

// ------------------------------------------ PAM q conv: M=16, K=128 -> qT[B][4096][32]
__global__ void __launch_bounds__(256) k_q_conv(
    const _Float16* __restrict__ A,       // [16][128]
    const _Float16* __restrict__ BT,      // [B][4096][128] (feat1T)
    const float* __restrict__ bias,       // [16]
    _Float16* __restrict__ qT) {          // [B][4096][32], cols 16..31 pre-zeroed
    const int tid = threadIdx.x, wv = tid >> 5, lane = tid & 31, nl = lane & 15, g = lane >> 4;
    const int b = blockIdx.y;
    const int pixbase = blockIdx.x * 128 + wv * 16;
    v8f acc = vzero();
    for (int k0 = 0; k0 < 128; k0 += 32) {
        v16h a = frag_a(A + (size_t)nl * 128 + k0, g);
        v16h bb = frag_b(BT + (((size_t)b << 12) + pixbase + nl) * 128 + k0, g);
        acc = wmma16(a, bb, acc);
    }
#pragma unroll
    for (int i = 0; i < 8; ++i) {
        int ck = g * 8 + i;
        int pix = pixbase + nl;
        qT[(((size_t)b << 12) + pix) * 32 + ck] = (_Float16)(acc[i] + bias[ck]);
    }
}

// ------------------------------------------ PAM flash attention (two pass)
// out[c,i] = gamma * (sum_j softmax_j(q_i.q_j) v[c,j]) + feat1[c,i]
__global__ void __launch_bounds__(256) k_pam(
    const _Float16* __restrict__ qT,   // [B][4096][32] (K padded with zeros)
    const _Float16* __restrict__ vh,   // [B][128][4096]
    const float* __restrict__ feat1,   // [B][128][4096]
    const float* __restrict__ gamma,
    _Float16* __restrict__ outh) {     // [B][128][4096]
    __shared__ _Float16 Ast[128 * 136];  // P tile [i][j], row stride 136 halves (272B)
    const int tid = threadIdx.x, wv = tid >> 5, lane = tid & 31, nl = lane & 15, g = lane >> 4;
    const int b = blockIdx.y;
    const int i0 = blockIdx.x * 128;
    const _Float16* Q = qT + ((size_t)b << 12) * 32;
    const _Float16* V = vh + (((size_t)b * 128) << 12);

    const v16h aq = frag_a(Q + (size_t)(i0 + wv * 16 + nl) * 32, g);

    // ---- pass 1: per-row running max & sum (rows: i = wv*16 + g*8 + ii)
    float rm[8], rs[8];
#pragma unroll
    for (int i = 0; i < 8; ++i) { rm[i] = -3.0e38f; rs[i] = 0.f; }

    for (int j0 = 0; j0 < 4096; j0 += 128) {
        v8f s[8];
#pragma unroll
        for (int nt = 0; nt < 8; ++nt) s[nt] = vzero();
#pragma unroll
        for (int nt = 0; nt < 8; ++nt) {
            v16h bb = frag_b(Q + (size_t)(j0 + nt * 16 + nl) * 32, g);
            s[nt] = wmma16(aq, bb, s[nt]);
        }
#pragma unroll
        for (int ii = 0; ii < 8; ++ii) {
            float lm = s[0][ii];
#pragma unroll
            for (int nt = 1; nt < 8; ++nt) lm = fmaxf(lm, s[nt][ii]);
#pragma unroll
            for (int msk = 1; msk < 16; msk <<= 1) lm = fmaxf(lm, __shfl_xor(lm, msk, 32));
            float nm = fmaxf(rm[ii], lm);
            float ls = 0.f;
#pragma unroll
            for (int nt = 0; nt < 8; ++nt) ls += __expf(s[nt][ii] - nm);
#pragma unroll
            for (int msk = 1; msk < 16; msk <<= 1) ls += __shfl_xor(ls, msk, 32);
            rs[ii] = rs[ii] * __expf(rm[ii] - nm) + ls;
            rm[ii] = nm;
        }
    }

    // ---- pass 2: O[i,c] = sum_j exp(S-m) * v[c,j]
    v8f o[8];
#pragma unroll
    for (int nt = 0; nt < 8; ++nt) o[nt] = vzero();

    for (int j0 = 0; j0 < 4096; j0 += 128) {
        v8f s[8];
#pragma unroll
        for (int nt = 0; nt < 8; ++nt) s[nt] = vzero();
#pragma unroll
        for (int nt = 0; nt < 8; ++nt) {
            v16h bb = frag_b(Q + (size_t)(j0 + nt * 16 + nl) * 32, g);
            s[nt] = wmma16(aq, bb, s[nt]);
        }
        __syncthreads();
#pragma unroll
        for (int nt = 0; nt < 8; ++nt) {
#pragma unroll
            for (int ii = 0; ii < 8; ++ii) {
                int row = wv * 16 + g * 8 + ii;
                Ast[row * 136 + nt * 16 + nl] = (_Float16)__expf(s[nt][ii] - rm[ii]);
            }
        }
        __syncthreads();
#pragma unroll
        for (int kc = 0; kc < 4; ++kc) {
            v16h pa = frag_a(&Ast[(wv * 16 + nl) * 136 + kc * 32], g);
#pragma unroll
            for (int nt = 0; nt < 8; ++nt) {
                v16h bb = frag_b(V + ((size_t)(nt * 16 + nl) << 12) + j0 + kc * 32, g);
                o[nt] = wmma16(pa, bb, o[nt]);
            }
        }
    }
    const float gm = gamma[0];
#pragma unroll
    for (int nt = 0; nt < 8; ++nt) {
#pragma unroll
        for (int ii = 0; ii < 8; ++ii) {
            int pix = i0 + wv * 16 + g * 8 + ii;
            int c = nt * 16 + nl;
            size_t idx = (((size_t)b * 128 + c) << 12) + pix;
            outh[idx] = (_Float16)(gm * (o[nt][ii] / rs[ii]) + feat1[idx]);
        }
    }
}

// ------------------------------------------ CAM energy: E[c,d] = sum_m f[c,m] f[d,m]
__global__ void __launch_bounds__(256) k_cam_energy(
    const _Float16* __restrict__ fh,      // [B][128][4096]
    float* __restrict__ energy) {         // [B][128][128]
    const int tid = threadIdx.x, wv = tid >> 5, lane = tid & 31, nl = lane & 15, g = lane >> 4;
    const int b = blockIdx.y;
    const _Float16* F = fh + (((size_t)b * 128) << 12);
    v8f acc[8];
#pragma unroll
    for (int nt = 0; nt < 8; ++nt) acc[nt] = vzero();
    for (int k0 = 0; k0 < 4096; k0 += 32) {
        v16h a = frag_a(F + ((size_t)(wv * 16 + nl) << 12) + k0, g);
#pragma unroll
        for (int nt = 0; nt < 8; ++nt) {
            v16h bb = frag_b(F + ((size_t)(nt * 16 + nl) << 12) + k0, g);
            acc[nt] = wmma16(a, bb, acc[nt]);
        }
    }
#pragma unroll
    for (int nt = 0; nt < 8; ++nt) {
#pragma unroll
        for (int i = 0; i < 8; ++i) {
            int c = wv * 16 + g * 8 + i;
            int d = nt * 16 + nl;
            energy[((size_t)b * 128 + c) * 128 + d] = acc[nt][i];
        }
    }
}

// attn[c,d] = softmax_d(max_d'(E) - E)  ==  exp(min_d'(E) - E_d) normalized
__global__ void k_cam_softmax(const float* __restrict__ energy, _Float16* __restrict__ attn) {
    const int b = blockIdx.y, c = threadIdx.x;  // 128 threads
    const float* E = energy + ((size_t)b * 128 + c) * 128;
    float mn = 3e38f;
    for (int d = 0; d < 128; ++d) mn = fminf(mn, E[d]);
    float sum = 0.f;
    for (int d = 0; d < 128; ++d) sum += __expf(mn - E[d]);
    float inv = 1.f / sum;
    _Float16* A = attn + ((size_t)b * 128 + c) * 128;
    for (int d = 0; d < 128; ++d) A[d] = (_Float16)(__expf(mn - E[d]) * inv);
}

// ============================================================== host launcher
extern "C" void kernel_launch(void* const* d_in, const int* in_sizes, int n_in,
                              void* d_out, int out_size, void* d_ws, size_t ws_size,
                              hipStream_t stream) {
    (void)in_sizes; (void)n_in; (void)out_size; (void)ws_size;
    const float* x       = (const float*)d_in[0];
    const float* w11     = (const float*)d_in[1];
    const float* g11     = (const float*)d_in[2];
    const float* b11     = (const float*)d_in[3];
    const float* m11     = (const float*)d_in[4];
    const float* v11     = (const float*)d_in[5];
    const float* w12     = (const float*)d_in[6];
    const float* g12     = (const float*)d_in[7];
    const float* b12     = (const float*)d_in[8];
    const float* m12     = (const float*)d_in[9];
    const float* v12     = (const float*)d_in[10];
    const float* pam_kw  = (const float*)d_in[11];
    const float* pam_kb  = (const float*)d_in[12];
    const float* pam_vw  = (const float*)d_in[13];
    const float* pam_vb  = (const float*)d_in[14];
    const float* pam_g   = (const float*)d_in[15];
    const float* cam_g   = (const float*)d_in[16];
    const float* w21     = (const float*)d_in[17];
    const float* g21     = (const float*)d_in[18];
    const float* b21     = (const float*)d_in[19];
    const float* m21     = (const float*)d_in[20];
    const float* v21     = (const float*)d_in[21];
    const float* w22     = (const float*)d_in[22];
    const float* g22     = (const float*)d_in[23];
    const float* b22     = (const float*)d_in[24];
    const float* m22     = (const float*)d_in[25];
    const float* v22     = (const float*)d_in[26];
    const float* w31     = (const float*)d_in[27];
    const float* b31     = (const float*)d_in[28];
    const float* w32     = (const float*)d_in[29];
    const float* b32     = (const float*)d_in[30];
    const float* w4      = (const float*)d_in[31];
    const float* b4      = (const float*)d_in[32];
    float* out = (float*)d_out;

    // ---- workspace bump allocator (256B aligned)
    char* wsp = (char*)d_ws;
    auto alloc = [&](size_t bytes) -> void* {
        void* r = (void*)wsp;
        wsp += (bytes + 255) & ~(size_t)255;
        return r;
    };
    const size_t NPIX = 4096, B = 4;
    _Float16* w11p   = (_Float16*)alloc(128 * 4608 * 2);
    _Float16* w12p   = (_Float16*)alloc(128 * 4608 * 2);
    _Float16* w21p   = (_Float16*)alloc(128 * 1152 * 2);
    _Float16* w22p   = (_Float16*)alloc(128 * 1152 * 2);
    _Float16* kwp    = (_Float16*)alloc(16 * 128 * 2);
    _Float16* vwp    = (_Float16*)alloc(128 * 128 * 2);
    _Float16* w31p   = (_Float16*)alloc(64 * 128 * 2);
    _Float16* w32p   = (_Float16*)alloc(64 * 128 * 2);
    _Float16* w4p    = (_Float16*)alloc(64 * 128 * 2);
    float* bnS       = (float*)alloc(4 * 128 * 4);
    float* bnB       = (float*)alloc(4 * 128 * 4);
    float* feat1_32  = (float*)alloc(B * 128 * NPIX * 4);
    _Float16* feat1T = (_Float16*)alloc(B * NPIX * 128 * 2);
    float* feat2_32  = (float*)alloc(B * 128 * NPIX * 4);
    _Float16* feat2h = (_Float16*)alloc(B * 128 * NPIX * 2);
    _Float16* feat2T = (_Float16*)alloc(B * NPIX * 128 * 2);
    _Float16* qTb    = (_Float16*)alloc(B * NPIX * 32 * 2);
    _Float16* v_h    = (_Float16*)alloc(B * 128 * NPIX * 2);
    _Float16* pa_h   = (_Float16*)alloc(B * 128 * NPIX * 2);
    _Float16* ca_h   = (_Float16*)alloc(B * 128 * NPIX * 2);
    _Float16* paCT   = (_Float16*)alloc(B * NPIX * 128 * 2);
    _Float16* caCT   = (_Float16*)alloc(B * NPIX * 128 * 2);
    _Float16* sumT   = (_Float16*)alloc(B * NPIX * 128 * 2);
    float* energy    = (float*)alloc(B * 128 * 128 * 4);
    _Float16* attn   = (_Float16*)alloc(B * 128 * 128 * 2);

    // ---- prep
    k_bn_prep<<<1, 128, 0, stream>>>(g11, b11, m11, v11, bnS + 0,   bnB + 0);
    k_bn_prep<<<1, 128, 0, stream>>>(g12, b12, m12, v12, bnS + 128, bnB + 128);
    k_bn_prep<<<1, 128, 0, stream>>>(g21, b21, m21, v21, bnS + 256, bnB + 256);
    k_bn_prep<<<1, 128, 0, stream>>>(g22, b22, m22, v22, bnS + 384, bnB + 384);
    k_pack_w3<<<256, 256, 0, stream>>>(w11, w11p, 512);
    k_pack_w3<<<256, 256, 0, stream>>>(w12, w12p, 512);
    k_pack_w3<<<64, 256, 0, stream>>>(w21, w21p, 128);
    k_pack_w3<<<64, 256, 0, stream>>>(w22, w22p, 128);
    k_cvt_h<<<8, 256, 0, stream>>>(pam_kw, kwp, 16 * 128);
    k_cvt_h<<<64, 256, 0, stream>>>(pam_vw, vwp, 128 * 128);
    k_cvt_h<<<32, 256, 0, stream>>>(w31, w31p, 64 * 128);
    k_cvt_h<<<32, 256, 0, stream>>>(w32, w32p, 64 * 128);
    k_cvt_h<<<32, 256, 0, stream>>>(w4, w4p, 64 * 128);
    k_zero_h<<<256, 256, 0, stream>>>(qTb, (int)(B * NPIX * 32));

    dim3 gNB(32, 4);

    // ---- feat1 / feat2 (conv3x3 512->128 + BN + ReLU)
    k_conv3x3<512, float><<<gNB, 256, 0, stream>>>(x, w11p, bnS + 0, bnB + 0,
                                                   feat1_32, nullptr, feat1T);
    k_conv3x3<512, float><<<gNB, 256, 0, stream>>>(x, w12p, bnS + 128, bnB + 128,
                                                   feat2_32, feat2h, feat2T);

    // ---- PAM
    k_q_conv<<<gNB, 256, 0, stream>>>(kwp, feat1T, pam_kb, qTb);
    k_gemm128<<<gNB, 256, 0, stream>>>(vwp, 0, feat1T, 128, pam_vb, nullptr, nullptr, v_h);
    k_pam<<<gNB, 256, 0, stream>>>(qTb, v_h, feat1_32, pam_g, pa_h);

    // ---- CAM
    k_cam_energy<<<dim3(1, 4), 256, 0, stream>>>(feat2h, energy);
    k_cam_softmax<<<dim3(1, 4), 128, 0, stream>>>(energy, attn);
    k_gemm128<<<gNB, 256, 0, stream>>>(attn, (size_t)128 * 128, feat2T, 128,
                                       nullptr, cam_g, feat2_32, ca_h);

    // ---- second convs (128->128 3x3 + BN + ReLU), transposed f16 outputs
    k_conv3x3<128, _Float16><<<gNB, 256, 0, stream>>>(pa_h, w21p, bnS + 256, bnB + 256,
                                                      nullptr, nullptr, paCT);
    k_conv3x3<128, _Float16><<<gNB, 256, 0, stream>>>(ca_h, w22p, bnS + 384, bnB + 384,
                                                      nullptr, nullptr, caCT);
    k_sum_h<<<256, 256, 0, stream>>>(paCT, caCT, sumT, (int)(B * NPIX * 128));

    // ---- 1x1 output convs -> d_out [3][B][64][4096]
    const size_t SEC = B * 64 * NPIX;
    k_gemm64<<<gNB, 256, 0, stream>>>(w31p, paCT, b31, out);
    k_gemm64<<<gNB, 256, 0, stream>>>(w32p, caCT, b32, out + SEC);
    k_gemm64<<<gNB, 256, 0, stream>>>(w4p, sumT, b4, out + 2 * SEC);
}
